// GCN3D_segR_4818953306460
// MI455X (gfx1250) — matrix operand order repaired and checked by reading, not compiled, not run on hardware
//
#include <hip/hip_runtime.h>
#include <hip/hip_bf16.h>

// ---------------------------------------------------------------------------
// Types for CDNA5 WMMA
// ---------------------------------------------------------------------------
typedef __bf16 bf16_t;
typedef __attribute__((ext_vector_type(16))) __bf16 v16bf;
typedef __attribute__((ext_vector_type(8)))  float  v8f;

#define BSZ   4
#define VNUM  2048
#define SNUM  7
#define KNBR  10

__device__ __forceinline__ bf16_t f2bf(float f) {
    union { float f; unsigned int u; } x; x.f = f;
    unsigned int r = x.u + 0x7FFFu + ((x.u >> 16) & 1u);   // RNE
    union { unsigned short s; bf16_t b; } y; y.s = (unsigned short)(r >> 16);
    return y.b;
}

// 16-half WMMA fragment viewed as two 16B vectors (forces b128 loads, no repack)
union frag16 {
    uint4 q[2];
    v16bf v;
};

// ---------------------------------------------------------------------------
// Register-blocked WMMA GEMM:
//   C[M,N] = A[M,K](bf16) * Bt[N,K]^T(bf16) + bias, optional ReLU.
// One wave computes a 64x32 macro-tile = 4(M) x 2(N) WMMA 16x16 tiles.
// Per K-step: 4 A-frags + 2 B-frags (12 x b128 loads) feed 8 WMMAs.
// Requires M % 64 == 0, N % 32 == 0, K % 32 == 0; all buffers 16B aligned,
// lda/ldb multiples of 8 elements (true for every call below).
// Fragment layouts follow CDNA5 ISA 7.12.2 (wave32).
// ---------------------------------------------------------------------------
__global__ void __launch_bounds__(32)
gemm_wmma_bf16(const bf16_t* __restrict__ A, int lda,
               const bf16_t* __restrict__ Bt, int ldb,
               float* __restrict__ C, int ldc,
               bf16_t* __restrict__ Cbf, int ldcb,
               const float* __restrict__ bias,
               int M, int N, int K, int Nvalid, int doRelu)
{
    const int tn   = blockIdx.x;          // 32-wide N tile
    const int tm   = blockIdx.y;          // 64-tall M tile
    const int lane = threadIdx.x;         // 0..31
    const int g    = lane >> 4;           // lane group 0/1
    const int r16  = lane & 15;

    const bf16_t* Arow0 = A  + ((size_t)tm * 64 + r16) * lda;  // A row M=r16 in subtile
    const bf16_t* Brow0 = Bt + ((size_t)tn * 32 + r16) * ldb;  // B col N=r16 in subtile

    v8f acc[4][2];
#pragma unroll
    for (int mi = 0; mi < 4; ++mi)
#pragma unroll
        for (int ni = 0; ni < 2; ++ni)
            acc[mi][ni] = (v8f){0.f, 0.f, 0.f, 0.f, 0.f, 0.f, 0.f, 0.f};

    for (int kk = 0; kk < K; kk += 32) {
        frag16 a[4], b[2];
        // A 16x32 bf16 frag: lane l holds M=l&15; halves 0..7 = K in [g*8,+8),
        //                    halves 8..15 = K in [16+g*8,+8)
#pragma unroll
        for (int mi = 0; mi < 4; ++mi) {
            const bf16_t* ap = Arow0 + (size_t)mi * 16 * lda + kk + g * 8;
            a[mi].q[0] = *(const uint4*)(ap);
            a[mi].q[1] = *(const uint4*)(ap + 16);
        }
        // B 32x16 bf16 frag: lane l holds N=l&15; halves 0..7 = K in [g*8,+8), halves 8..15 = K in [16+g*8,+8)
#pragma unroll
        for (int ni = 0; ni < 2; ++ni) {
            const bf16_t* bp = Brow0 + (size_t)ni * 16 * ldb + kk + g * 8;
            b[ni].q[0] = *(const uint4*)(bp);
            b[ni].q[1] = *(const uint4*)(bp + 16);
        }
        // Speculative near-cache prefetch of next K step (dropped if OOB).
        __builtin_prefetch(Arow0 + kk + 32, 0, 3);
        __builtin_prefetch(Brow0 + kk + 32, 0, 3);
#pragma unroll
        for (int mi = 0; mi < 4; ++mi)
#pragma unroll
            for (int ni = 0; ni < 2; ++ni)
                acc[mi][ni] = __builtin_amdgcn_wmma_f32_16x16x32_bf16(
                    false, a[mi].v, false, b[ni].v, (short)0, acc[mi][ni], false, false);
    }

    // C/D 16x16 f32: VGPR r -> M = g*8 + r, lane -> N = r16
#pragma unroll
    for (int ni = 0; ni < 2; ++ni) {
        const int n = tn * 32 + ni * 16 + r16;
        if (n < Nvalid) {
            const float bv = bias ? bias[n] : 0.f;
#pragma unroll
            for (int mi = 0; mi < 4; ++mi) {
#pragma unroll
                for (int r = 0; r < 8; ++r) {
                    const int m = tm * 64 + mi * 16 + g * 8 + r;
                    float v = acc[mi][ni][r] + bv;
                    if (doRelu) v = fmaxf(v, 0.f);
                    if (C)   C  [(size_t)m * ldc  + n] = v;
                    if (Cbf) Cbf[(size_t)m * ldcb + n] = f2bf(v);
                }
            }
        }
    }
}

// ---------------------------------------------------------------------------
// Helper / elementwise kernels
// ---------------------------------------------------------------------------
__global__ void normalize_cols_kernel(const float* __restrict__ w,
                                      float* __restrict__ out, int ncols)
{
    int j = blockIdx.x * blockDim.x + threadIdx.x;
    if (j >= ncols) return;
    float a = w[j], b = w[ncols + j], c = w[2 * ncols + j];
    float n = fmaxf(sqrtf(a * a + b * b + c * c), 1e-12f);
    out[j] = a / n; out[ncols + j] = b / n; out[2 * ncols + j] = c / n;
}

__global__ void to_bf16_kernel(const float* __restrict__ x,
                               bf16_t* __restrict__ y, int n)
{
    int t = blockIdx.x * blockDim.x + threadIdx.x;
    if (t < n) y[t] = f2bf(x[t]);
}

// w[K][N] row-major -> out[N][K] bf16
__global__ void wT_bf16_kernel(const float* __restrict__ w,
                               bf16_t* __restrict__ out, int K, int N)
{
    int t = blockIdx.x * blockDim.x + threadIdx.x;
    if (t >= N * K) return;
    int n = t / K, k = t % K;
    out[t] = f2bf(w[(size_t)k * N + n]);
}

// w[Nin][Kin] row-major -> out[Nout][Kpad] bf16, zero padded
__global__ void w_pad_bf16_kernel(const float* __restrict__ w,
                                  bf16_t* __restrict__ out,
                                  int Nout, int Nin, int Kin, int Kpad)
{
    int t = blockIdx.x * blockDim.x + threadIdx.x;
    if (t >= Nout * Kpad) return;
    int n = t / Kpad, k = t % Kpad;
    out[t] = (n < Nin && k < Kin) ? f2bf(w[(size_t)n * Kin + k]) : f2bf(0.f);
}

// ---------------------------------------------------------------------------
// kNN (self-excluded, matches top_k(-dist)[...,1:]) — one thread per query
// ---------------------------------------------------------------------------
__global__ void knn_kernel(const float* __restrict__ vq, const float* __restrict__ vs,
                           int B, int Vq, int Vs, int qStride, int sStride,
                           int Kn, int* __restrict__ idx, int selfExcl)
{
    int t = blockIdx.x * blockDim.x + threadIdx.x;
    if (t >= B * Vq) return;
    int b = t / Vq, q = t % Vq;
    const float* qp = vq + ((size_t)b * qStride + q) * 3;
    float qx = qp[0], qy = qp[1], qz = qp[2];
    float bd[KNBR]; int bi[KNBR];
    for (int i = 0; i < Kn; ++i) { bd[i] = 3.4e38f; bi[i] = 0; }
    const float* sp = vs + (size_t)b * sStride * 3;
    for (int m = 0; m < Vs; ++m) {
        if (selfExcl && m == q) continue;
        float dx = sp[3 * m]     - qx;
        float dy = sp[3 * m + 1] - qy;
        float dz = sp[3 * m + 2] - qz;
        float d = dx * dx + dy * dy + dz * dz;
        if (d < bd[Kn - 1]) {
            int j = Kn - 1;
            while (j > 0 && d < bd[j - 1]) { bd[j] = bd[j - 1]; bi[j] = bi[j - 1]; --j; }
            bd[j] = d; bi[j] = m;
        }
    }
    for (int i = 0; i < Kn; ++i) idx[(size_t)t * Kn + i] = bi[i];
}

__global__ void nearest_kernel(const float* __restrict__ vq, const float* __restrict__ vs,
                               int B, int Vq, int Vs, int qStride, int sStride,
                               int* __restrict__ out)
{
    int t = blockIdx.x * blockDim.x + threadIdx.x;
    if (t >= B * Vq) return;
    int b = t / Vq, q = t % Vq;
    const float* qp = vq + ((size_t)b * qStride + q) * 3;
    float qx = qp[0], qy = qp[1], qz = qp[2];
    const float* sp = vs + (size_t)b * sStride * 3;
    float best = 3.4e38f; int bestI = 0;
    for (int m = 0; m < Vs; ++m) {
        float dx = sp[3 * m] - qx, dy = sp[3 * m + 1] - qy, dz = sp[3 * m + 2] - qz;
        float d = dx * dx + dy * dy + dz * dz;
        if (d < best) { best = d; bestI = m; }
    }
    out[t] = bestI;
}

// ---------------------------------------------------------------------------
// conv_surface: out[b,v,c] = relu( sum_s max_n relu(nd_n . sd[:, s*C+c]) )
// block = C threads, one block per (b,v)
// ---------------------------------------------------------------------------
__global__ void conv_surface_kernel(const int* __restrict__ nb,
                                    const float* __restrict__ verts,
                                    const float* __restrict__ sd,
                                    float* __restrict__ out,
                                    int B, int Vq, int Kn, int Sn, int outc)
{
    __shared__ float nd[KNBR][3];
    const int bv = blockIdx.x;
    const int b = bv / Vq, v = bv % Vq;
    if (threadIdx.x < (unsigned)Kn) {
        int j = nb[(size_t)bv * Kn + threadIdx.x];
        const float* pv = verts + ((size_t)b * Vq + v) * 3;
        const float* pj = verts + ((size_t)b * Vq + j) * 3;
        float dx = pj[0] - pv[0], dy = pj[1] - pv[1], dz = pj[2] - pv[2];
        float n = fmaxf(sqrtf(dx * dx + dy * dy + dz * dz), 1e-12f);
        nd[threadIdx.x][0] = dx / n; nd[threadIdx.x][1] = dy / n; nd[threadIdx.x][2] = dz / n;
    }
    __syncthreads();
    const int c = threadIdx.x;
    const int scols = Sn * outc;
    float acc = 0.f;
    for (int s = 0; s < Sn; ++s) {
        int col = s * outc + c;
        float d0 = sd[col], d1 = sd[scols + col], d2 = sd[2 * scols + col];
        float best = 0.f;
        for (int n = 0; n < Kn; ++n) {
            float th = fmaxf(nd[n][0] * d0 + nd[n][1] * d1 + nd[n][2] * d2, 0.f);
            best = fmaxf(best, th);
        }
        acc += best;
    }
    out[(size_t)bv * outc + c] = fmaxf(acc, 0.f);
}

// ---------------------------------------------------------------------------
// conv_layer combine: out = fo[:,:C] + sum_s max_n( relu(nd.sd) * fo[nb, C+s*C+c] )
// ---------------------------------------------------------------------------
__global__ void conv_combine_kernel(const int* __restrict__ nb,
                                    const float* __restrict__ verts, int vStride,
                                    const float* __restrict__ fo,
                                    const float* __restrict__ sd,
                                    float* __restrict__ out,
                                    int B, int Vq, int Kn, int Sn, int outc)
{
    __shared__ float nd[KNBR][3];
    __shared__ int   nbs[KNBR];
    const int bv = blockIdx.x;
    const int b = bv / Vq, v = bv % Vq;
    if (threadIdx.x < (unsigned)Kn) {
        int j = nb[(size_t)bv * Kn + threadIdx.x];
        nbs[threadIdx.x] = j;
        const float* pv = verts + ((size_t)b * vStride + v) * 3;
        const float* pj = verts + ((size_t)b * vStride + j) * 3;
        float dx = pj[0] - pv[0], dy = pj[1] - pv[1], dz = pj[2] - pv[2];
        float n = fmaxf(sqrtf(dx * dx + dy * dy + dz * dz), 1e-12f);
        nd[threadIdx.x][0] = dx / n; nd[threadIdx.x][1] = dy / n; nd[threadIdx.x][2] = dz / n;
    }
    __syncthreads();
    const int c = threadIdx.x;
    const int scols = Sn * outc;
    const int foW = (Sn + 1) * outc;
    float acc = fo[(size_t)bv * foW + c];       // center
    for (int s = 0; s < Sn; ++s) {
        int col = s * outc + c;
        float d0 = sd[col], d1 = sd[scols + col], d2 = sd[2 * scols + col];
        float best = -3.4e38f;
        for (int n = 0; n < Kn; ++n) {
            float th = fmaxf(nd[n][0] * d0 + nd[n][1] * d1 + nd[n][2] * d2, 0.f);
            float supp = fo[((size_t)b * Vq + nbs[n]) * foW + outc + col];
            best = fmaxf(best, th * supp);
        }
        acc += best;
    }
    out[(size_t)bv * outc + c] = acc;
}

// ---------------------------------------------------------------------------
// Train-mode BN (+ReLU) per channel: one block per channel
// ---------------------------------------------------------------------------
__global__ void bn_relu_kernel(const float* __restrict__ x,
                               const float* __restrict__ g,
                               const float* __restrict__ be,
                               float* __restrict__ y, int rows, int C)
{
    __shared__ float s1[256], s2[256];
    const int c = blockIdx.x;
    float s = 0.f, sq = 0.f;
    for (int r = threadIdx.x; r < rows; r += 256) {
        float v = x[(size_t)r * C + c];
        s += v; sq += v * v;
    }
    s1[threadIdx.x] = s; s2[threadIdx.x] = sq;
    __syncthreads();
    for (int off = 128; off > 0; off >>= 1) {
        if (threadIdx.x < (unsigned)off) {
            s1[threadIdx.x] += s1[threadIdx.x + off];
            s2[threadIdx.x] += s2[threadIdx.x + off];
        }
        __syncthreads();
    }
    const float mean = s1[0] / rows;
    const float var  = s2[0] / rows - mean * mean;
    const float scale = rsqrtf(var + 1e-5f) * g[c];
    const float shift = be[c] - mean * scale;
    for (int r = threadIdx.x; r < rows; r += 256) {
        float v = x[(size_t)r * C + c] * scale + shift;
        y[(size_t)r * C + c] = fmaxf(v, 0.f);
    }
}

__global__ void pool_max_kernel(const float* __restrict__ fm, int Vin,
                                const int* __restrict__ idx,
                                float* __restrict__ out,
                                int B, int Vq, int Kn, int C)
{
    int t = blockIdx.x * blockDim.x + threadIdx.x;
    if (t >= B * Vq * C) return;
    int c = t % C; int bv = t / C;
    int b = bv / Vq, v = bv % Vq;
    float best = -3.4e38f;
    for (int n = 0; n < Kn; ++n) {
        int j = idx[((size_t)b * Vq + v) * Kn + n];
        best = fmaxf(best, fm[((size_t)b * Vin + j) * C + c]);
    }
    out[(size_t)bv * C + c] = best;
}

__global__ void copy_subverts_kernel(const float* __restrict__ src, int srcStride,
                                     float* __restrict__ dst, int B, int Vq)
{
    int t = blockIdx.x * blockDim.x + threadIdx.x;
    if (t >= B * Vq * 3) return;
    int b = t / (Vq * 3); int rem = t % (Vq * 3);
    int v = rem / 3, d = rem % 3;
    dst[t] = src[((size_t)b * srcStride + v) * 3 + d];
}

__global__ void global_max_kernel(const float* __restrict__ fm,
                                  float* __restrict__ out, int B, int Vr, int C)
{
    int t = blockIdx.x * blockDim.x + threadIdx.x;
    if (t >= B * C) return;
    int b = t / C, c = t % C;
    float best = -3.4e38f;
    for (int r = 0; r < Vr; ++r)
        best = fmaxf(best, fm[((size_t)b * Vr + r) * C + c]);
    out[t] = best;
}

// ---------------------------------------------------------------------------
// Build fuse (bf16 GEMM-A, padded to 1824) and feat output (f32) in one pass
// ---------------------------------------------------------------------------
__global__ void assemble_fuse_kernel(const float* __restrict__ fm0,
                                     const float* __restrict__ fm1,
                                     const float* __restrict__ fm2,
                                     const float* __restrict__ fm3,
                                     const float* __restrict__ fm4,
                                     const float* __restrict__ fg,
                                     const float* __restrict__ onehot,
                                     const int* __restrict__ n1,
                                     const int* __restrict__ n2,
                                     bf16_t* __restrict__ fuseA,
                                     float* __restrict__ feat,
                                     int B, int V)
{
    const int bv = blockIdx.x;
    const int b = bv / V;
    const int i1 = n1[bv];
    const int i2 = n2[bv];
    for (int c = threadIdx.x; c < 1824; c += blockDim.x) {
        float val; int fc = -1;
        if      (c < 128)  { val = fm0[(size_t)bv * 128 + c];                     fc = c; }
        else if (c < 256)  { val = fm1[(size_t)bv * 128 + (c - 128)];             fc = c; }
        else if (c < 512)  { val = fm2[((size_t)b * 512 + i1) * 256 + (c - 256)]; fc = c; }
        else if (c < 768)  { val = fm3[((size_t)b * 512 + i1) * 256 + (c - 512)]; fc = c; }
        else if (c < 1280) { val = fm4[((size_t)b * 128 + i2) * 512 + (c - 768)]; fc = c; }
        else if (c < 1792) { val = fg[(size_t)b * 512 + (c - 1280)]; }
        else if (c < 1808) { val = onehot[(size_t)b * 16 + (c - 1792)]; fc = 1280 + (c - 1792); }
        else               { val = 0.f; }
        fuseA[(size_t)bv * 1824 + c] = f2bf(val);
        if (fc >= 0) feat[(size_t)bv * 1296 + fc] = val;
    }
}

__global__ void split_pred_kernel(const float* __restrict__ pred,
                                  float* __restrict__ seg,
                                  float* __restrict__ vec, int rows)
{
    int t = blockIdx.x * blockDim.x + threadIdx.x;
    if (t >= rows * 30) return;
    int r = t / 30, j = t % 30;
    float v = pred[(size_t)r * 32 + j];
    if (j < 6) seg[(size_t)r * 6 + j] = v;
    else       vec[(size_t)r * 24 + (j - 6)] = v;
}

// ---------------------------------------------------------------------------
// Host launcher
// ---------------------------------------------------------------------------
extern "C" void kernel_launch(void* const* d_in, const int* in_sizes, int n_in,
                              void* d_out, int out_size, void* d_ws, size_t ws_size,
                              hipStream_t stream)
{
    const float* verts = (const float*)d_in[0];
    const float* onehot= (const float*)d_in[1];
    const float* dir0  = (const float*)d_in[2];
    const float* w1    = (const float*)d_in[3];
    const float* b1    = (const float*)d_in[4];
    const float* dir1  = (const float*)d_in[5];
    const float* w2    = (const float*)d_in[6];
    const float* b2    = (const float*)d_in[7];
    const float* dir2  = (const float*)d_in[8];
    const float* w3    = (const float*)d_in[9];
    const float* b3    = (const float*)d_in[10];
    const float* dir3  = (const float*)d_in[11];
    const float* w4    = (const float*)d_in[12];
    const float* b4    = (const float*)d_in[13];
    const float* dir4  = (const float*)d_in[14];
    const float* g1    = (const float*)d_in[15];
    const float* be1   = (const float*)d_in[16];
    const float* g2    = (const float*)d_in[17];
    const float* be2   = (const float*)d_in[18];
    const float* g3    = (const float*)d_in[19];
    const float* be3   = (const float*)d_in[20];
    const float* wc1   = (const float*)d_in[21];
    const float* bc1   = (const float*)d_in[22];
    const float* wc2   = (const float*)d_in[23];
    const float* bc2   = (const float*)d_in[24];
    const float* wc3   = (const float*)d_in[25];
    const float* bc3   = (const float*)d_in[26];
    (void)in_sizes; (void)n_in; (void)out_size; (void)ws_size;

    float* out    = (float*)d_out;
    float* outSeg = out;                       // [8192, 6]
    float* outVec = out + (size_t)8192 * 6;    // [8192, 24]
    float* outFeat= out + (size_t)8192 * 30;   // [8192, 1296]

    char* p = (char*)d_ws;
    auto alloc = [&](size_t bytes) -> void* {
        void* r = (void*)p; p += (bytes + 255) & ~(size_t)255; return r;
    };

    const int BV = BSZ * VNUM;                 // 8192

    float* sd0 = (float*)alloc(3 * 896  * 4);
    float* sd1 = (float*)alloc(3 * 896  * 4);
    float* sd2 = (float*)alloc(3 * 1792 * 4);
    float* sd3 = (float*)alloc(3 * 1792 * 4);
    float* sd4 = (float*)alloc(3 * 3584 * 4);
    int*   nb0 = (int*)  alloc((size_t)BV * KNBR * 4);
    float* fm0 = (float*)alloc((size_t)BV * 128 * 4);
    bf16_t* a0 = (bf16_t*)alloc((size_t)BV * 128 * 2);
    bf16_t* w1T= (bf16_t*)alloc((size_t)1024 * 128 * 2);
    float* fo  = (float*)alloc((size_t)BV * 1024 * 4);   // reused fo1..fo4
    float* cv  = (float*)alloc((size_t)BV * 128 * 4);    // reused cv1..cv3
    float* fm1 = (float*)alloc((size_t)BV * 128 * 4);
    int*  idxp1= (int*)  alloc((size_t)BSZ * 512 * 4 * 4);
    float* vp1 = (float*)alloc((size_t)BSZ * 512 * 3 * 4);
    float* fp1 = (float*)alloc((size_t)BSZ * 512 * 128 * 4);
    int*   nb1 = (int*)  alloc((size_t)BSZ * 512 * KNBR * 4);
    bf16_t* a1 = (bf16_t*)alloc((size_t)2048 * 128 * 2);
    bf16_t* w2T= (bf16_t*)alloc((size_t)2048 * 128 * 2);
    float* fm2 = (float*)alloc((size_t)2048 * 256 * 4);
    bf16_t* a2 = (bf16_t*)alloc((size_t)2048 * 256 * 2);
    bf16_t* w3T= (bf16_t*)alloc((size_t)2048 * 256 * 2);
    float* fm3 = (float*)alloc((size_t)2048 * 256 * 4);
    int*  idxp2= (int*)  alloc((size_t)BSZ * 128 * 4 * 4);
    float* vp2 = (float*)alloc((size_t)BSZ * 128 * 3 * 4);
    float* fp2 = (float*)alloc((size_t)BSZ * 128 * 256 * 4);
    int*   nb2 = (int*)  alloc((size_t)BSZ * 128 * KNBR * 4);
    bf16_t* a3 = (bf16_t*)alloc((size_t)512 * 256 * 2);
    bf16_t* w4T= (bf16_t*)alloc((size_t)4096 * 256 * 2);
    float* fm4 = (float*)alloc((size_t)512 * 512 * 4);
    float* fg  = (float*)alloc((size_t)BSZ * 512 * 4);
    int*   n1  = (int*)  alloc((size_t)BV * 4);
    int*   n2  = (int*)  alloc((size_t)BV * 4);
    bf16_t* fuseA = (bf16_t*)alloc((size_t)BV * 1824 * 2);
    bf16_t* wc1T  = (bf16_t*)alloc((size_t)512 * 1824 * 2);
    bf16_t* h1    = (bf16_t*)alloc((size_t)BV * 512 * 2);
    bf16_t* wc2T  = (bf16_t*)alloc((size_t)512 * 512 * 2);
    bf16_t* h2    = (bf16_t*)alloc((size_t)BV * 512 * 2);
    bf16_t* wc3T  = (bf16_t*)alloc((size_t)32 * 512 * 2);
    float* pred   = (float*)alloc((size_t)BV * 32 * 4);

    // --- normalized direction banks ---
    normalize_cols_kernel<<<(896 + 255) / 256, 256, 0, stream>>>(dir0, sd0, 896);
    normalize_cols_kernel<<<(896 + 255) / 256, 256, 0, stream>>>(dir1, sd1, 896);
    normalize_cols_kernel<<<(1792 + 255) / 256, 256, 0, stream>>>(dir2, sd2, 1792);
    normalize_cols_kernel<<<(1792 + 255) / 256, 256, 0, stream>>>(dir3, sd3, 1792);
    normalize_cols_kernel<<<(3584 + 255) / 256, 256, 0, stream>>>(dir4, sd4, 3584);

    // --- level 0 graph + surface conv ---
    knn_kernel<<<(BV + 63) / 64, 64, 0, stream>>>(verts, verts, BSZ, VNUM, VNUM,
                                                  VNUM, VNUM, KNBR, nb0, 1);
    conv_surface_kernel<<<BV, 128, 0, stream>>>(nb0, verts, sd0, fm0,
                                                BSZ, VNUM, KNBR, SNUM, 128);

    // --- conv layer 1 (out 128) ---
    to_bf16_kernel<<<(BV * 128 + 255) / 256, 256, 0, stream>>>(fm0, a0, BV * 128);
    wT_bf16_kernel<<<(1024 * 128 + 255) / 256, 256, 0, stream>>>(w1, w1T, 128, 1024);
    gemm_wmma_bf16<<<dim3(1024 / 32, BV / 64), 32, 0, stream>>>(
        a0, 128, w1T, 128, fo, 1024, nullptr, 0, b1, BV, 1024, 128, 1024, 0);
    conv_combine_kernel<<<BV, 128, 0, stream>>>(nb0, verts, VNUM, fo, sd1, cv,
                                                BSZ, VNUM, KNBR, SNUM, 128);
    bn_relu_kernel<<<128, 256, 0, stream>>>(cv, g1, be1, fm1, BV, 128);

    // --- pool 1 (2048 -> 512) ---
    knn_kernel<<<(BSZ * 512 + 63) / 64, 64, 0, stream>>>(verts, verts, BSZ, 512, VNUM,
                                                         VNUM, VNUM, 4, idxp1, 1);
    copy_subverts_kernel<<<(BSZ * 512 * 3 + 255) / 256, 256, 0, stream>>>(verts, VNUM, vp1, BSZ, 512);
    pool_max_kernel<<<(BSZ * 512 * 128 + 255) / 256, 256, 0, stream>>>(
        fm1, VNUM, idxp1, fp1, BSZ, 512, 4, 128);
    knn_kernel<<<(BSZ * 512 + 63) / 64, 64, 0, stream>>>(vp1, vp1, BSZ, 512, 512,
                                                         512, 512, KNBR, nb1, 1);

    // --- conv layer 2 (out 256) ---
    to_bf16_kernel<<<(2048 * 128 + 255) / 256, 256, 0, stream>>>(fp1, a1, 2048 * 128);
    wT_bf16_kernel<<<(2048 * 128 + 255) / 256, 256, 0, stream>>>(w2, w2T, 128, 2048);
    gemm_wmma_bf16<<<dim3(2048 / 32, 2048 / 64), 32, 0, stream>>>(
        a1, 128, w2T, 128, fo, 2048, nullptr, 0, b2, 2048, 2048, 128, 2048, 0);
    conv_combine_kernel<<<2048, 256, 0, stream>>>(nb1, vp1, 512, fo, sd2, cv,
                                                  BSZ, 512, KNBR, SNUM, 256);
    bn_relu_kernel<<<256, 256, 0, stream>>>(cv, g2, be2, fm2, 2048, 256);

    // --- conv layer 3 (out 256) ---
    to_bf16_kernel<<<(2048 * 256 + 255) / 256, 256, 0, stream>>>(fm2, a2, 2048 * 256);
    wT_bf16_kernel<<<(2048 * 256 + 255) / 256, 256, 0, stream>>>(w3, w3T, 256, 2048);
    gemm_wmma_bf16<<<dim3(2048 / 32, 2048 / 64), 32, 0, stream>>>(
        a2, 256, w3T, 256, fo, 2048, nullptr, 0, b3, 2048, 2048, 256, 2048, 0);
    conv_combine_kernel<<<2048, 256, 0, stream>>>(nb1, vp1, 512, fo, sd3, cv,
                                                  BSZ, 512, KNBR, SNUM, 256);
    bn_relu_kernel<<<256, 256, 0, stream>>>(cv, g3, be3, fm3, 2048, 256);

    // --- pool 2 (512 -> 128) ---
    knn_kernel<<<(BSZ * 128 + 63) / 64, 64, 0, stream>>>(vp1, vp1, BSZ, 128, 512,
                                                         512, 512, 4, idxp2, 1);
    copy_subverts_kernel<<<(BSZ * 128 * 3 + 255) / 256, 256, 0, stream>>>(vp1, 512, vp2, BSZ, 128);
    pool_max_kernel<<<(BSZ * 128 * 256 + 255) / 256, 256, 0, stream>>>(
        fm3, 512, idxp2, fp2, BSZ, 128, 4, 256);
    knn_kernel<<<(BSZ * 128 + 63) / 64, 64, 0, stream>>>(vp2, vp2, BSZ, 128, 128,
                                                         128, 128, KNBR, nb2, 1);

    // --- conv layer 4 (out 512, no BN/ReLU) ---
    to_bf16_kernel<<<(512 * 256 + 255) / 256, 256, 0, stream>>>(fp2, a3, 512 * 256);
    wT_bf16_kernel<<<(4096 * 256 + 255) / 256, 256, 0, stream>>>(w4, w4T, 256, 4096);
    gemm_wmma_bf16<<<dim3(4096 / 32, 512 / 64), 32, 0, stream>>>(
        a3, 256, w4T, 256, fo, 4096, nullptr, 0, b4, 512, 4096, 256, 4096, 0);
    conv_combine_kernel<<<512, 512, 0, stream>>>(nb2, vp2, 128, fo, sd4, fm4,
                                                 BSZ, 128, KNBR, SNUM, 512);

    // --- global feature, upsample indices ---
    global_max_kernel<<<(BSZ * 512 + 255) / 256, 256, 0, stream>>>(fm4, fg, BSZ, 128, 512);
    nearest_kernel<<<(BV + 63) / 64, 64, 0, stream>>>(verts, vp1, BSZ, VNUM, 512,
                                                      VNUM, 512, n1);
    nearest_kernel<<<(BV + 63) / 64, 64, 0, stream>>>(verts, vp2, BSZ, VNUM, 128,
                                                      VNUM, 128, n2);

    // --- fuse assembly (also writes feat output) ---
    assemble_fuse_kernel<<<BV, 128, 0, stream>>>(fm0, fm1, fm2, fm3, fm4, fg,
                                                 onehot, n1, n2, fuseA, outFeat,
                                                 BSZ, VNUM);

    // --- MLP head ---
    w_pad_bf16_kernel<<<(512 * 1824 + 255) / 256, 256, 0, stream>>>(wc1, wc1T, 512, 512, 1808, 1824);
    gemm_wmma_bf16<<<dim3(512 / 32, BV / 64), 32, 0, stream>>>(
        fuseA, 1824, wc1T, 1824, nullptr, 0, h1, 512, bc1, BV, 512, 1824, 512, 1);
    w_pad_bf16_kernel<<<(512 * 512 + 255) / 256, 256, 0, stream>>>(wc2, wc2T, 512, 512, 512, 512);
    gemm_wmma_bf16<<<dim3(512 / 32, BV / 64), 32, 0, stream>>>(
        h1, 512, wc2T, 512, nullptr, 0, h2, 512, bc2, BV, 512, 512, 512, 1);
    w_pad_bf16_kernel<<<(32 * 512 + 255) / 256, 256, 0, stream>>>(wc3, wc3T, 32, 30, 512, 512);
    gemm_wmma_bf16<<<dim3(32 / 32, BV / 64), 32, 0, stream>>>(
        h2, 512, wc3T, 512, pred, 32, nullptr, 0, bc3, BV, 32, 512, 30, 0);
    split_pred_kernel<<<(BV * 30 + 255) / 256, 256, 0, stream>>>(pred, outSeg, outVec, BV);
}
